// LoRAExpert_17849884082531
// MI455X (gfx1250) — matrix operand order repaired and checked by reading, not compiled, not run on hardware
//
#include <hip/hip_runtime.h>
#include <hip/hip_bf16.h>

// gfx1250 wave32 WMMA types
typedef __attribute__((ext_vector_type(16))) __bf16 v16bf;
typedef __attribute__((ext_vector_type(8)))  float  v8f;

#define LDA 40   // padded LDS row stride (bf16 elems): 80B, 16B-aligned, conflict-free

__device__ __forceinline__ int expert_of_row(const int* __restrict__ gs, int row, int nE) {
  int cum = 0, e = nE - 1;
  for (int i = 0; i < nE; ++i) {
    int nxt = cum + gs[i];
    if (row < nxt) { e = i; break; }
    cum = nxt;
  }
  return e;
}

// A-fragment (16x32 bf16, MxK): lane = M row; elems 0..7 = K+(kh?8:0)+0..7,
// elems 8..15 = K+16+(kh?8:0)+0..7  (kh = lane>=16)
__device__ __forceinline__ v16bf frag_a16(const __bf16* base, int row, int kh) {
  union { v16bf v; uint4 u[2]; } f;
  const __bf16* p = base + row * LDA + kh * 8;
  f.u[0] = *reinterpret_cast<const uint4*>(p);
  f.u[1] = *reinterpret_cast<const uint4*>(p + 16);
  return f.v;
}
// B-fragment (32x16 bf16, KxN) from a K-contiguous (transposed) LDS tile:
// lane = N col; elems 0..7 = K+(kh?8:0)+0..7, elems 8..15 = K+16+(kh?8:0)+0..7
__device__ __forceinline__ v16bf frag_b16(const __bf16* base, int col, int kh) {
  union { v16bf v; uint4 u[2]; } f;
  const __bf16* p = base + col * LDA + kh * 8;
  f.u[0] = *reinterpret_cast<const uint4*>(p);
  f.u[1] = *reinterpret_cast<const uint4*>(p + 16);
  return f.v;
}

// ---------------------------------------------------------------------------
// Kernel 1: inter[t][r] = x[t] . lora_A[a_t, e_t][:, r]   (bf16 WMMA, adapter-masked)
// Block: 256 threads = 8 waves, 256 token rows per block, N = R = 16.
// Software-pipelined: next K tile loaded into registers during WMMA burst.
// ---------------------------------------------------------------------------
__global__ __launch_bounds__(256)
void lora_inter_kernel(const float* __restrict__ x, const float* __restrict__ loraA,
                       const int* __restrict__ gsz, const int* __restrict__ aidx,
                       float* __restrict__ inter) {
  __shared__ alignas(16) __bf16 sX[256 * LDA];
  __shared__ alignas(16) __bf16 sA[4][16 * LDA];   // [adapter][r][k] (K-contiguous)

  const int tid  = threadIdx.x;
  const int lane = tid & 31, wave = tid >> 5;
  const int l16  = lane & 15, kh = lane >> 4;
  const int r0   = blockIdx.x * 256;
  const int e    = expert_of_row(gsz, r0, 8);

  int adv[2];
  #pragma unroll
  for (int mt = 0; mt < 2; ++mt) adv[mt] = aidx[r0 + wave * 32 + mt * 16 + l16];

  v8f acc[2];
  #pragma unroll
  for (int mt = 0; mt < 2; ++mt)
    for (int i = 0; i < 8; ++i) acc[mt][i] = 0.f;

  // lora_A staging split: 64 threads per adapter, 2 threads per k-row of 16 r's
  const int la_a = tid >> 6;
  const int la_t = tid & 63;
  const int la_k = la_t >> 1;
  const int la_r = (la_t & 1) * 8;

  union { v16bf v; uint4 u[2]; } zf;
  zf.u[0] = make_uint4(0u, 0u, 0u, 0u);
  zf.u[1] = make_uint4(0u, 0u, 0u, 0u);

  float4 rx[8], ra[2];
  auto load_tile = [&](int k0) {
    const float* gx = x + (size_t)(r0 + tid) * 1024 + k0;
    #pragma unroll
    for (int i = 0; i < 8; ++i) rx[i] = *reinterpret_cast<const float4*>(gx + 4 * i);
    const float* ga = loraA + ((size_t)(la_a * 8 + e) * 1024 + k0 + la_k) * 16 + la_r;
    #pragma unroll
    for (int i = 0; i < 2; ++i) ra[i] = *reinterpret_cast<const float4*>(ga + 4 * i);
  };

  load_tile(0);
  for (int k0 = 0; k0 < 1024; k0 += 32) {
    // stage current registers -> LDS (f32 -> bf16)
    #pragma unroll
    for (int i = 0; i < 8; ++i) {
      const float* pv = reinterpret_cast<const float*>(&rx[i]);
      #pragma unroll
      for (int j = 0; j < 4; ++j) sX[tid * LDA + 4 * i + j] = (__bf16)pv[j];
    }
    #pragma unroll
    for (int i = 0; i < 2; ++i) {
      const float* pv = reinterpret_cast<const float*>(&ra[i]);
      #pragma unroll
      for (int j = 0; j < 4; ++j) sA[la_a][(la_r + 4 * i + j) * LDA + la_k] = (__bf16)pv[j];
    }
    __syncthreads();

    if (k0 + 32 < 1024) load_tile(k0 + 32);   // overlap next loads with WMMAs

    v16bf af[2];
    #pragma unroll
    for (int mt = 0; mt < 2; ++mt) af[mt] = frag_a16(sX, wave * 32 + mt * 16 + l16, kh);

    #pragma unroll
    for (int a = 0; a < 4; ++a) {
      v16bf bfr = frag_b16(sA[a], l16, kh);
      #pragma unroll
      for (int mt = 0; mt < 2; ++mt) {
        v16bf am = (adv[mt] == a) ? af[mt] : zf.v;   // adapter mask (per-lane row)
        acc[mt] = __builtin_amdgcn_wmma_f32_16x16x32_bf16(
            false, am, false, bfr, (short)0, acc[mt], false, false);
      }
    }
    __syncthreads();
  }

  #pragma unroll
  for (int mt = 0; mt < 2; ++mt) {
    int rowb = r0 + wave * 32 + mt * 16 + kh * 8;
    #pragma unroll
    for (int v = 0; v < 8; ++v)
      inter[(size_t)(rowb + v) * 16 + l16] = acc[mt][v];
  }
}

// ---------------------------------------------------------------------------
// Kernel 2: out = x . W[e]  (bf16 hi/lo 3-term fp32 emulation)  +  fused
//           s[a] * inter . B[a,e]  epilogue (zero-padded K=16 WMMA, masked)
// Block: 256 threads = 8 waves; 128x128 output tile; wave = 32x64 (2x4 tiles).
// Register-double-buffered K loop; epilogue B staged through (dead) main LDS.
// ---------------------------------------------------------------------------
__global__ __launch_bounds__(256)
void lora_moe_main(const float* __restrict__ x, const float* __restrict__ weight,
                   const float* __restrict__ loraB, const float* __restrict__ scal,
                   const int* __restrict__ gsz, const int* __restrict__ aidx,
                   const float* __restrict__ inter, float* __restrict__ out) {
  __shared__ alignas(16) __bf16 smem[4][128 * LDA];   // xh / xl / wTh / wTl, reused in epilogue
  __bf16* sXh = smem[0];
  __bf16* sXl = smem[1];
  __bf16* sWh = smem[2];
  __bf16* sWl = smem[3];

  const int tid  = threadIdx.x;
  const int lane = tid & 31, wave = tid >> 5;
  const int l16  = lane & 15, kh = lane >> 4;
  const int wm   = wave & 3;        // 4 row groups of 32
  const int wn   = wave >> 2;       // 2 col groups of 64
  const int r0   = blockIdx.x * 128;
  const int c0   = blockIdx.y * 128;

  const int e = expert_of_row(gsz, r0, 8);
  const float* wexp = weight + (size_t)e * 1024 * 1024;

  // x stage:  thread -> row (0..127), 16-col half
  const int xrow = tid >> 1,       xcb = (tid & 1) * 16;
  // w stage:  thread -> 4 consecutive k rows x 4 cols (enables packed b64 LDS stores)
  const int wkg  = (tid >> 5) * 4, wnb = (tid & 31) * 4;

  v8f acc[2][4];
  #pragma unroll
  for (int mt = 0; mt < 2; ++mt)
    for (int nt = 0; nt < 4; ++nt)
      for (int i = 0; i < 8; ++i) acc[mt][nt][i] = 0.f;

  float4 rx[4], rw[4];
  auto load_tile = [&](int k0) {
    const float* gx = x + (size_t)(r0 + xrow) * 1024 + k0 + xcb;
    #pragma unroll
    for (int i = 0; i < 4; ++i) rx[i] = *reinterpret_cast<const float4*>(gx + 4 * i);
    #pragma unroll
    for (int r = 0; r < 4; ++r)
      rw[r] = *reinterpret_cast<const float4*>(wexp + (size_t)(k0 + wkg + r) * 1024 + c0 + wnb);
  };

  load_tile(0);
  for (int k0 = 0; k0 < 1024; k0 += 32) {
    // ---- stage x hi/lo (contiguous -> compiler packs ds_store_b128) ----
    #pragma unroll
    for (int i = 0; i < 4; ++i) {
      const float* pv = reinterpret_cast<const float*>(&rx[i]);
      #pragma unroll
      for (int j = 0; j < 4; ++j) {
        float val = pv[j];
        __bf16 h = (__bf16)val;
        __bf16 l = (__bf16)(val - (float)h);
        sXh[xrow * LDA + xcb + 4 * i + j] = h;
        sXl[xrow * LDA + xcb + 4 * i + j] = l;
      }
    }
    // ---- stage W transposed hi/lo, packed 4-along-K -> ds_store_b64 ----
    #pragma unroll
    for (int j = 0; j < 4; ++j) {
      union { __bf16 h[4]; uint2 u; } ph, pl;
      #pragma unroll
      for (int r = 0; r < 4; ++r) {
        float val = reinterpret_cast<const float*>(&rw[r])[j];
        __bf16 h = (__bf16)val;
        ph.h[r] = h;
        pl.h[r] = (__bf16)(val - (float)h);
      }
      *reinterpret_cast<uint2*>(&sWh[(wnb + j) * LDA + wkg]) = ph.u;
      *reinterpret_cast<uint2*>(&sWl[(wnb + j) * LDA + wkg]) = pl.u;
    }
    __syncthreads();

    if (k0 + 32 < 1024) load_tile(k0 + 32);   // next tile in flight during WMMAs

    // ---- 3-term split-precision compute: xh*wh + xh*wl + xl*wh ----
    v16bf ah[2], al[2];
    #pragma unroll
    for (int mt = 0; mt < 2; ++mt) {
      ah[mt] = frag_a16(sXh, wm * 32 + mt * 16 + l16, kh);
      al[mt] = frag_a16(sXl, wm * 32 + mt * 16 + l16, kh);
    }
    #pragma unroll
    for (int nt = 0; nt < 4; ++nt) {
      v16bf bh = frag_b16(sWh, wn * 64 + nt * 16 + l16, kh);
      v16bf bl = frag_b16(sWl, wn * 64 + nt * 16 + l16, kh);
      #pragma unroll
      for (int mt = 0; mt < 2; ++mt) {
        acc[mt][nt] = __builtin_amdgcn_wmma_f32_16x16x32_bf16(
            false, al[mt], false, bh, (short)0, acc[mt][nt], false, false);
        acc[mt][nt] = __builtin_amdgcn_wmma_f32_16x16x32_bf16(
            false, ah[mt], false, bl, (short)0, acc[mt][nt], false, false);
        acc[mt][nt] = __builtin_amdgcn_wmma_f32_16x16x32_bf16(
            false, ah[mt], false, bh, (short)0, acc[mt][nt], false, false);
      }
    }
    __syncthreads();
  }

  // ---- stage lora_B[a,e] 16x128 slices into (now dead) LDS, transposed+zero-padded ----
  {
    const int br = tid >> 4;            // r = 0..15
    const int bc = (tid & 15) * 8;      // col base 0..120
    #pragma unroll
    for (int a = 0; a < 4; ++a) {
      const float* bp = loraB + ((size_t)(a * 8 + e) * 16 + br) * 1024 + c0 + bc;
      float4 v0 = *reinterpret_cast<const float4*>(bp);
      float4 v1 = *reinterpret_cast<const float4*>(bp + 4);
      const float* p0 = reinterpret_cast<const float*>(&v0);
      const float* p1 = reinterpret_cast<const float*>(&v1);
      #pragma unroll
      for (int j = 0; j < 4; ++j) {
        smem[a][(bc + j) * LDA + br]     = (__bf16)p0[j];
        smem[a][(bc + 4 + j) * LDA + br] = (__bf16)p1[j];
        smem[a][(bc + j) * LDA + 16 + br]     = (__bf16)0.f;  // K=16..31 zero pad
        smem[a][(bc + 4 + j) * LDA + 16 + br] = (__bf16)0.f;
      }
    }
  }

  // adapter index, intermediate activations, scalings
  float sc[4];
  #pragma unroll
  for (int a = 0; a < 4; ++a) sc[a] = scal[a];
  int adv[2]; float iv[2][8];
  #pragma unroll
  for (int mt = 0; mt < 2; ++mt) {
    int row = r0 + wm * 32 + mt * 16 + l16;
    adv[mt] = aidx[row];
    const float* ip = inter + (size_t)row * 16 + kh * 8;
    #pragma unroll
    for (int j = 0; j < 8; ++j) iv[mt][j] = ip[j];
  }
  __syncthreads();

  // ---- fused LoRA-B epilogue: acc += s[a] * maskA(inter) . B[a,e] ----
  #pragma unroll
  for (int a = 0; a < 4; ++a) {
    union { v16bf v; uint4 u[2]; } la[2];
    #pragma unroll
    for (int mt = 0; mt < 2; ++mt) {
      float f = (adv[mt] == a) ? sc[a] : 0.f;   // mask + scaling folded into A frag
      #pragma unroll
      for (int j = 0; j < 8; ++j) {
        la[mt].v[j]     = (__bf16)(iv[mt][j] * f);
        la[mt].v[8 + j] = (__bf16)0.f;          // K = 16..31 zero padding
      }
    }
    #pragma unroll
    for (int nt = 0; nt < 4; ++nt) {
      v16bf lb = frag_b16(smem[a], wn * 64 + nt * 16 + l16, kh);
      #pragma unroll
      for (int mt = 0; mt < 2; ++mt)
        acc[mt][nt] = __builtin_amdgcn_wmma_f32_16x16x32_bf16(
            false, la[mt].v, false, lb, (short)0, acc[mt][nt], false, false);
    }
  }

  // ---- store (D layout: m = vgpr + 8*(lane>=16), n = lane%16) ----
  #pragma unroll
  for (int mt = 0; mt < 2; ++mt) {
    int rowb = r0 + wm * 32 + mt * 16 + kh * 8;
    #pragma unroll
    for (int nt = 0; nt < 4; ++nt) {
      int col = c0 + wn * 64 + nt * 16 + l16;
      #pragma unroll
      for (int v = 0; v < 8; ++v)
        out[(size_t)(rowb + v) * 1024 + col] = acc[mt][nt][v];
    }
  }
}

// ---------------------------------------------------------------------------
extern "C" void kernel_launch(void* const* d_in, const int* in_sizes, int n_in,
                              void* d_out, int out_size, void* d_ws, size_t ws_size,
                              hipStream_t stream) {
  (void)n_in; (void)ws_size;
  const float* x        = (const float*)d_in[0];
  const float* weight   = (const float*)d_in[1];
  const float* lora_A   = (const float*)d_in[2];
  const float* lora_B   = (const float*)d_in[3];
  const float* lora_s   = (const float*)d_in[4];
  const int*   gsz      = (const int*)d_in[5];
  const int*   aidx     = (const int*)d_in[6];
  float*       out      = (float*)d_out;
  float*       inter    = (float*)d_ws;              // T x 16 f32 scratch (512 KB)

  const int T    = in_sizes[0] / 1024;
  const int DOUT = out_size / T;

  lora_inter_kernel<<<dim3(T / 256), dim3(256), 0, stream>>>(x, lora_A, gsz, aidx, inter);
  lora_moe_main<<<dim3(T / 128, DOUT / 128), dim3(256), 0, stream>>>(
      x, weight, lora_B, lora_s, gsz, aidx, inter, out);
}
